// StableLMAttention_61873298866850
// MI455X (gfx1250) — compile-verified, matrix operand order checked by reading, not run on hardware
//
#include <hip/hip_runtime.h>
#include <hip/hip_bf16.h>

// ---------------------------------------------------------------------------
// StableLM attention block for MI455X (gfx1250, wave32, WMMA bf16).
//   B=2, S=2048, H=2048, NH=32, NKV=8, D=64, ROT=16
// Pipeline:
//   1) fp32 -> bf16 conversion of hidden, w_qkv, w_o
//   2) QKV GEMM  (bf16 WMMA, fp32 acc, double-buffered TDM/async LDS staging)
//   3) per-head LN + partial NeoX RoPE (fp32), scatter Q/K bf16, V^T bf16
//   4) flash attention (bf16 WMMA QK^T and PV, fp32 online softmax)
//   5) output GEMM (bf16 WMMA, fp32 out)
// ---------------------------------------------------------------------------

typedef __attribute__((ext_vector_type(16))) __bf16 v16bf;
typedef __attribute__((ext_vector_type(8)))  __bf16 v8bf;
typedef __attribute__((ext_vector_type(8)))  float  v8f;
typedef int v4i_gcc __attribute__((vector_size(16)));   // async builtin pointee

#define N_HEADS 32
#define N_KV    8
#define HEAD_D  64
#define SEQ     2048
#define BATCH   2
#define HID     2048
#define QKV_N   3072
#define TOKENS  (BATCH * SEQ)   // 4096

// ---------------- gfx1250 async global->LDS (ASYNCcnt) support -------------
#if defined(__HIP_DEVICE_COMPILE__) && defined(__has_builtin)
#if __has_builtin(__builtin_amdgcn_global_load_async_to_lds_b128)
#define USE_ASYNC_LDS 1
#else
#define USE_ASYNC_LDS 0
#endif
#if __has_builtin(__builtin_amdgcn_tensor_load_to_lds)
#define USE_TDM 1
#else
#define USE_TDM 0
#pragma message("gfx1250: tensor_load_to_lds builtin MISSING - async staging")
#endif
#else
#define USE_ASYNC_LDS 0
#define USE_TDM 0
#endif

__device__ __forceinline__ void async_ld16(const __bf16* g, __bf16* l) {
#if USE_ASYNC_LDS
  __builtin_amdgcn_global_load_async_to_lds_b128(
      (__attribute__((address_space(1))) v4i_gcc*)g,
      (__attribute__((address_space(3))) v4i_gcc*)l, 0, 0);
#else
  *(v8bf*)l = *(const v8bf*)g;
#endif
}

template <int N>
__device__ __forceinline__ void wait_async() {
#if USE_ASYNC_LDS
#if __has_builtin(__builtin_amdgcn_s_wait_asynccnt)
  __builtin_amdgcn_s_wait_asynccnt(N);
#else
  asm volatile("s_wait_asynccnt %0" ::"i"(N) : "memory");
#endif
#endif
}

#if USE_TDM
typedef unsigned int u32x4 __attribute__((vector_size(16)));
typedef int i32x8 __attribute__((vector_size(32)));
typedef int i32x4 __attribute__((vector_size(16)));

// Build a 2-D D# (groups 0/1; groups 2/3 zero) and issue TENSOR_LOAD_TO_LDS.
// This toolchain's builtin is the 6-arg form:
//   (uint32x4 g0, int32x8 g1, int32x4 g2, int32x4 g3, int32x8 pad, i32 cpol)
//   gtile         : global address of tile start (bf16 elements, row-major)
//   lds_byte_addr : LDS destination byte address
//   tensor_w/h    : tensor extent (elements / rows) for OOB clamping
//   tile_w/h      : tile extent   (elements / rows)
//   row_stride    : elements between consecutive rows
__device__ __forceinline__ void tdm_load_tile_2d(
    const __bf16* gtile, unsigned lds_byte_addr,
    unsigned tensor_w, unsigned tensor_h,
    unsigned tile_w, unsigned tile_h, unsigned row_stride) {
  unsigned long long ga = (unsigned long long)(size_t)gtile;
  u32x4 g0;
  g0[0] = 1u;                                             // count=1, user D#
  g0[1] = lds_byte_addr;                                  // bits  63:32
  g0[2] = (unsigned)ga;                                   // bits  95:64
  g0[3] = ((unsigned)(ga >> 32) & 0x01ffffffu)            // bits 120:96
        | (2u << 30);                                     // type=2 (image)
  i32x8 g1;
  g1[0] = 1 << 16;                                        // data_size=2B
  g1[1] = (int)((tensor_w & 0xffffu) << 16);              // tensor_dim0 lo
  g1[2] = (int)(((tensor_w >> 16) & 0xffffu) |            // tensor_dim0 hi
                ((tensor_h & 0xffffu) << 16));            // tensor_dim1 lo
  g1[3] = (int)(((tensor_h >> 16) & 0xffffu) |            // tensor_dim1 hi
                ((tile_w & 0xffffu) << 16));              // tile_dim0
  g1[4] = (int)(tile_h & 0xffffu);                        // tile_dim1 (dim2=0)
  g1[5] = (int)row_stride;                                // dim0_stride lo
  g1[6] = 0;                                              // stride hi
  g1[7] = 0;
  i32x4 z4 = (i32x4){0, 0, 0, 0};
  i32x8 z8 = (i32x8){0, 0, 0, 0, 0, 0, 0, 0};
  __builtin_amdgcn_tensor_load_to_lds(g0, g1, z4, z4, z8, 0);
}

__device__ __forceinline__ unsigned lds_addr_of(const void* p) {
  return (unsigned)(size_t)(__attribute__((address_space(3))) const void*)p;
}
#endif  // USE_TDM

__device__ __forceinline__ void wait_tensor0() {
#if defined(__HIP_DEVICE_COMPILE__) && __has_builtin(__builtin_amdgcn_s_wait_tensorcnt)
  __builtin_amdgcn_s_wait_tensorcnt(0);
#else
  asm volatile("s_wait_tensorcnt 0x0" ::: "memory");
#endif
}

// ------------------------------ fp32 -> bf16 -------------------------------
__global__ void cvt_f32_bf16(const float* __restrict__ in,
                             __bf16* __restrict__ out, int n) {
  int i = blockIdx.x * blockDim.x + threadIdx.x;
  if (i < n) out[i] = (__bf16)in[i];
}

// ------------------------- WMMA bf16 GEMM: C = A * B^T ---------------------
// A: [M,K] bf16 row-major, B: [N,K] bf16 row-major (K contiguous for both),
// C: [M,N] fp32. Block tile 128x128, 8 waves, wave tile 32x64.
// Double-buffered: TDM (or async engine) fills tile k+1 while WMMAs consume k.
#define BM 128
#define BN 128
#define BK 32

__global__ __launch_bounds__(256)
void gemm_bf16_wmma(const __bf16* __restrict__ A,
                    const __bf16* __restrict__ B,
                    float* __restrict__ C,
                    int M, int N, int K) {
  __shared__ __bf16 sA[2][BM][BK];   // 2 x 8 KB
  __shared__ __bf16 sB[2][BN][BK];   // 2 x 8 KB

  const int tid  = threadIdx.x;
  const int wave = tid >> 5;      // 0..7
  const int lane = tid & 31;
  const int wm   = wave & 3;      // 4 wave rows of 32
  const int wn   = wave >> 2;     // 2 wave cols of 64
  const int lrow = lane & 15;
  const int lhk  = (lane >> 4) * 8;     // K sub-pattern select (wave32 layout)

  const int block_m = blockIdx.y * BM;
  const int block_n = blockIdx.x * BN;

  v8f acc[2][4];
#pragma unroll
  for (int i = 0; i < 2; ++i)
#pragma unroll
    for (int j = 0; j < 4; ++j) acc[i][j] = (v8f)0.0f;

#if !USE_TDM
  // each thread moves two 16B chunks of A and two of B per K-tile
  const int c0 = tid,       r0 = c0 >> 2, cc0 = (c0 & 3) * 8;
  const int c1 = tid + 256, r1 = c1 >> 2, cc1 = (c1 & 3) * 8;
  auto stage = [&](int k0, int b) {
    async_ld16(&A[(size_t)(block_m + r0) * K + k0 + cc0], &sA[b][r0][cc0]);
    async_ld16(&B[(size_t)(block_n + r0) * K + k0 + cc0], &sB[b][r0][cc0]);
    async_ld16(&A[(size_t)(block_m + r1) * K + k0 + cc1], &sA[b][r1][cc1]);
    async_ld16(&B[(size_t)(block_n + r1) * K + k0 + cc1], &sB[b][r1][cc1]);
  };
#endif

  int buf = 0;
#if USE_TDM
  if (wave == 0) {   // one wave drives the Tensor Data Mover for the block
    tdm_load_tile_2d(&A[(size_t)block_m * K], lds_addr_of(&sA[0][0][0]),
                     K, BM, BK, BM, K);
    tdm_load_tile_2d(&B[(size_t)block_n * K], lds_addr_of(&sB[0][0][0]),
                     K, BN, BK, BN, K);
  }
#else
  stage(0, 0);
#endif

  for (int k0 = 0; k0 < K; k0 += BK) {
#if USE_TDM
    if (wave == 0) wait_tensor0();   // current buffer's 2 descriptors done
    __syncthreads();                 // release tile; prior readers of buf^1 done
    if (wave == 0 && k0 + BK < K) {
      tdm_load_tile_2d(&A[(size_t)block_m * K + k0 + BK],
                       lds_addr_of(&sA[buf ^ 1][0][0]), K, BM, BK, BM, K);
      tdm_load_tile_2d(&B[(size_t)block_n * K + k0 + BK],
                       lds_addr_of(&sB[buf ^ 1][0][0]), K, BN, BK, BN, K);
    }
#else
    if (k0 + BK < K) {
      stage(k0 + BK, buf ^ 1);   // prefetch next tile (4 async ops)
      wait_async<4>();           // in-order: first 4 (current tile) done
    } else {
      wait_async<0>();
    }
    __syncthreads();
#endif

    v16bf afrag[2], bfrag[4];
#pragma unroll
    for (int i = 0; i < 2; ++i) {
      int row = wm * 32 + i * 16 + lrow;
      *((v8bf*)&afrag[i])       = *(const v8bf*)&sA[buf][row][lhk];
      *(((v8bf*)&afrag[i]) + 1) = *(const v8bf*)&sA[buf][row][lhk + 16];
    }
#pragma unroll
    for (int j = 0; j < 4; ++j) {
      int col = wn * 64 + j * 16 + lrow;
      *((v8bf*)&bfrag[j])       = *(const v8bf*)&sB[buf][col][lhk];
      *(((v8bf*)&bfrag[j]) + 1) = *(const v8bf*)&sB[buf][col][lhk + 16];
    }
#pragma unroll
    for (int i = 0; i < 2; ++i)
#pragma unroll
      for (int j = 0; j < 4; ++j)
        acc[i][j] = __builtin_amdgcn_wmma_f32_16x16x32_bf16(
            false, afrag[i], false, bfrag[j], (short)0, acc[i][j],
            false, false);
#if !USE_TDM
    __syncthreads();   // all waves done reading `buf` before it is refilled
#endif
    buf ^= 1;
  }

  // store: C layout -> VGPR r holds M = r + (lane>=16 ? 8 : 0), N = lane&15
#pragma unroll
  for (int i = 0; i < 2; ++i)
#pragma unroll
    for (int j = 0; j < 4; ++j)
#pragma unroll
      for (int r = 0; r < 8; ++r) {
        int m = block_m + wm * 32 + i * 16 + (lane >> 4) * 8 + r;
        int n = block_n + wn * 64 + j * 16 + (lane & 15);
        C[(size_t)m * N + n] = acc[i][j][r];
      }
}

// ----------------- per-head LayerNorm + partial NeoX RoPE ------------------
// grid (40, 4096): blockIdx.x = head slot (0..31 Q, 32..39 K), .y = token
__global__ __launch_bounds__(64)
void ln_rope_kernel(const float* __restrict__ qkv,
                    const int* __restrict__ pos_ids,
                    const float* __restrict__ q_ln_w,
                    const float* __restrict__ k_ln_w,
                    __bf16* __restrict__ Qo,    // [B][NH][S][D]
                    __bf16* __restrict__ Ko) {  // [B][NKV][S][D]
  const int hh = blockIdx.x;
  const int t  = blockIdx.y;
  const int d  = threadIdx.x;
  const bool isq = hh < N_HEADS;
  const int head = isq ? hh : hh - N_HEADS;
  const size_t base =
      (size_t)t * QKV_N + (isq ? head * HEAD_D : HID + head * HEAD_D);

  float x = qkv[base + d];
  __shared__ float red[64];
  __shared__ float sy[64];

  red[d] = x; __syncthreads();
  for (int s = 32; s > 0; s >>= 1) { if (d < s) red[d] += red[d + s]; __syncthreads(); }
  float mu = red[0] * (1.0f / 64.0f);
  __syncthreads();
  float c = x - mu;
  red[d] = c * c; __syncthreads();
  for (int s = 32; s > 0; s >>= 1) { if (d < s) red[d] += red[d + s]; __syncthreads(); }
  float var = red[0] * (1.0f / 64.0f);

  const float* w = isq ? q_ln_w : k_ln_w;
  float y = c * rsqrtf(var + 1e-5f) * w[head * HEAD_D + d];

  sy[d] = y; __syncthreads();
  float outv = y;
  if (d < 16) {               // rotary on first 16 dims, NeoX half-split
    int   pos = pos_ids[t];
    int   i   = d & 7;
    float inv = __powf(10000.0f, -(float)i * 0.125f);
    float fr  = (float)pos * inv;
    float cs  = __cosf(fr), sn = __sinf(fr);
    float x1 = sy[i], x2 = sy[i + 8];
    outv = (d < 8) ? (x1 * cs - x2 * sn) : (x2 * cs + x1 * sn);
  }
  int b = t >> 11, s = t & (SEQ - 1);
  if (isq)
    Qo[(((size_t)(b * N_HEADS + head) * SEQ + s) << 6) + d] = (__bf16)outv;
  else
    Ko[(((size_t)(b * N_KV + head) * SEQ + s) << 6) + d] = (__bf16)outv;
}

// ---------------- V slice -> transposed bf16 [B][NKV][D][S] ----------------
__global__ void vcopy_kernel(const float* __restrict__ qkv,
                             __bf16* __restrict__ vT) {
  int idx = blockIdx.x * blockDim.x + threadIdx.x;
  if (idx >= BATCH * N_KV * SEQ * HEAD_D) return;
  int d  = idx & 63;
  int kv = (idx >> 6) & 7;
  int t  = idx >> 9;          // token
  float v = qkv[(size_t)t * QKV_N + HID + N_KV * HEAD_D + kv * HEAD_D + d];
  int b = t >> 11, s = t & (SEQ - 1);
  vT[((size_t)(b * N_KV + kv) * HEAD_D + d) * SEQ + s] = (__bf16)v;
}

// ------------------------- flash attention (WMMA) --------------------------
// grid (16 qtiles, 32 heads, 2 batch), 256 threads = 8 waves.
// Each wave owns 16 query rows; streams keys in tiles of 32.
__global__ __launch_bounds__(256)
void attn_kernel(const __bf16* __restrict__ Q,   // [B][NH][S][D]
                 const __bf16* __restrict__ K,   // [B][NKV][S][D]
                 const __bf16* __restrict__ VT,  // [B][NKV][D][S]
                 __bf16* __restrict__ O) {       // [B][S][NH*D]
  const int b = blockIdx.z, h = blockIdx.y;
  const int kvh = h / (N_HEADS / N_KV);
  const int tid = threadIdx.x, wave = tid >> 5, lane = tid & 31;
  const int lrow = lane & 15;
  const int lhk  = (lane >> 4) * 8;

  const __bf16* Qh = Q + (size_t)(b * N_HEADS + h) * SEQ * HEAD_D;
  const __bf16* Kh = K + (size_t)(b * N_KV + kvh) * SEQ * HEAD_D;
  const __bf16* Vh = VT + (size_t)(b * N_KV + kvh) * HEAD_D * SEQ;

  const int q0 = blockIdx.x * 128 + wave * 16;   // wave's first query row

  // Q fragments: 16 rows x 64 dims -> two A-frags (K=0..31, 32..63)
  v16bf qf[2];
#pragma unroll
  for (int i = 0; i < 2; ++i) {
    int row = q0 + lrow;
    *((v8bf*)&qf[i])       = *(const v8bf*)&Qh[(size_t)row * HEAD_D + i * 32 + lhk];
    *(((v8bf*)&qf[i]) + 1) = *(const v8bf*)&Qh[(size_t)row * HEAD_D + i * 32 + lhk + 16];
  }

  float m_i[8], l_i[8];
  v8f acc[4];
#pragma unroll
  for (int r = 0; r < 8; ++r) { m_i[r] = -1e30f; l_i[r] = 0.0f; }
#pragma unroll
  for (int c = 0; c < 4; ++c) acc[c] = (v8f)0.0f;

  __shared__ __bf16 sP[8][16][32];   // per-wave P tile slab (8 KB)

  const float scale = 0.125f;        // 1/sqrt(64)
  const int mbase = q0 + (lane >> 4) * 8;
  const int kend = q0 + 16;          // causal horizon (exclusive)

  for (int k0 = 0; k0 < kend; k0 += 32) {
    // prefetch next key/value tiles into cache (global_prefetch_b8 path)
    if (k0 + 32 < kend) {
      __builtin_prefetch(&Kh[(size_t)(k0 + 32 + lrow) * HEAD_D], 0, 1);
      __builtin_prefetch(&Vh[(size_t)lrow * SEQ + k0 + 32], 0, 1);
    }
    // ---- scores S = Q K^T : 16x32 in two 16x16 C-frags --------------------
    v8f sc[2];
#pragma unroll
    for (int j = 0; j < 2; ++j) sc[j] = (v8f)0.0f;
#pragma unroll
    for (int j = 0; j < 2; ++j) {
      int key = k0 + j * 16 + lrow;
#pragma unroll
      for (int i = 0; i < 2; ++i) {
        v16bf kf;
        *((v8bf*)&kf)       = *(const v8bf*)&Kh[(size_t)key * HEAD_D + i * 32 + lhk];
        *(((v8bf*)&kf) + 1) = *(const v8bf*)&Kh[(size_t)key * HEAD_D + i * 32 + lhk + 16];
        sc[j] = __builtin_amdgcn_wmma_f32_16x16x32_bf16(
            false, qf[i], false, kf, (short)0, sc[j], false, false);
      }
    }
    // ---- scale + causal mask ---------------------------------------------
#pragma unroll
    for (int j = 0; j < 2; ++j) {
      int n = k0 + j * 16 + (lane & 15);
#pragma unroll
      for (int r = 0; r < 8; ++r) {
        float s = sc[j][r] * scale;
        sc[j][r] = (n <= mbase + r) ? s : -1e30f;
      }
    }
    // ---- online softmax: row reductions across the 16-lane row group -----
    float mnew[8], corr[8];
#pragma unroll
    for (int r = 0; r < 8; ++r) {
      float mx = fmaxf(sc[0][r], sc[1][r]);
#pragma unroll
      for (int off = 1; off < 16; off <<= 1)
        mx = fmaxf(mx, __shfl_xor(mx, off, 32));
      mnew[r] = fmaxf(m_i[r], mx);
      corr[r] = __expf(m_i[r] - mnew[r]);
      m_i[r]  = mnew[r];
      l_i[r] *= corr[r];
    }
#pragma unroll
    for (int j = 0; j < 2; ++j)
#pragma unroll
      for (int r = 0; r < 8; ++r)
        sc[j][r] = __expf(sc[j][r] - mnew[r]);
#pragma unroll
    for (int r = 0; r < 8; ++r) {
      float sm = sc[0][r] + sc[1][r];
#pragma unroll
      for (int off = 1; off < 16; off <<= 1)
        sm += __shfl_xor(sm, off, 32);
      l_i[r] += sm;
    }
#pragma unroll
    for (int c = 0; c < 4; ++c)
#pragma unroll
      for (int r = 0; r < 8; ++r) acc[c][r] *= corr[r];

    // ---- P: C-layout -> LDS -> A-frag layout ------------------------------
#pragma unroll
    for (int j = 0; j < 2; ++j)
#pragma unroll
      for (int r = 0; r < 8; ++r)
        sP[wave][(lane >> 4) * 8 + r][j * 16 + (lane & 15)] = (__bf16)sc[j][r];
    // wave-internal LDS RAW: DS ops are in-order per wave, just drain DScnt
    asm volatile("s_wait_dscnt 0x0" ::: "memory");

    v16bf pf;
    *((v8bf*)&pf)       = *(const v8bf*)&sP[wave][lrow][lhk];
    *(((v8bf*)&pf) + 1) = *(const v8bf*)&sP[wave][lrow][lhk + 16];

    // ---- O += P V : V^T gives contiguous B-frag loads ---------------------
#pragma unroll
    for (int c = 0; c < 4; ++c) {
      const __bf16* Vc = Vh + (size_t)(c * 16 + lrow) * SEQ;
      v16bf vf;
      *((v8bf*)&vf)       = *(const v8bf*)&Vc[k0 + lhk];
      *(((v8bf*)&vf) + 1) = *(const v8bf*)&Vc[k0 + lhk + 16];
      acc[c] = __builtin_amdgcn_wmma_f32_16x16x32_bf16(
          false, pf, false, vf, (short)0, acc[c], false, false);
    }
  }

  // ---- epilogue: normalize by l_i, store token-major bf16 -----------------
#pragma unroll
  for (int c = 0; c < 4; ++c)
#pragma unroll
    for (int r = 0; r < 8; ++r) {
      int m = q0 + (lane >> 4) * 8 + r;
      int d = c * 16 + (lane & 15);
      float o = acc[c][r] / l_i[r];
      O[((size_t)(b * SEQ + m)) * HID + h * HEAD_D + d] = (__bf16)o;
    }
}

// ---------------------------------------------------------------------------
extern "C" void kernel_launch(void* const* d_in, const int* in_sizes, int n_in,
                              void* d_out, int out_size, void* d_ws,
                              size_t ws_size, hipStream_t stream) {
  const int*   pos  = (const int*)  d_in[0];
  const float* hid  = (const float*)d_in[1];
  const float* wqkv = (const float*)d_in[2];
  const float* qlnw = (const float*)d_in[3];
  const float* klnw = (const float*)d_in[4];
  const float* wo   = (const float*)d_in[5];
  float* out = (float*)d_out;

  char* ws = (char*)d_ws;
  size_t off = 0;
  auto carve = [&](size_t bytes) -> char* {
    char* p = ws + off;
    off += (bytes + 255) & ~(size_t)255;
    return p;
  };
  __bf16* hb    = (__bf16*)carve((size_t)TOKENS * HID * 2);          // 16 MB
  __bf16* wqb   = (__bf16*)carve((size_t)QKV_N * HID * 2);           // 12 MB
  __bf16* wob   = (__bf16*)carve((size_t)HID * HID * 2);             //  8 MB
  float*  qkv32 = (float*) carve((size_t)TOKENS * QKV_N * 4);        // 48 MB
  __bf16* qb    = (__bf16*)carve((size_t)BATCH * N_HEADS * SEQ * HEAD_D * 2);
  __bf16* kb    = (__bf16*)carve((size_t)BATCH * N_KV * SEQ * HEAD_D * 2);
  __bf16* vT    = (__bf16*)carve((size_t)BATCH * N_KV * SEQ * HEAD_D * 2);
  __bf16* ao    = (__bf16*)carve((size_t)TOKENS * HID * 2);          // 16 MB
  (void)ws_size; (void)in_sizes; (void)n_in; (void)out_size;

  // 1) fp32 -> bf16
  {
    int n1 = TOKENS * HID;          // 8388608
    int n2 = QKV_N * HID;           // 6291456
    int n3 = HID * HID;             // 4194304
    cvt_f32_bf16<<<(n1 + 255) / 256, 256, 0, stream>>>(hid, hb, n1);
    cvt_f32_bf16<<<(n2 + 255) / 256, 256, 0, stream>>>(wqkv, wqb, n2);
    cvt_f32_bf16<<<(n3 + 255) / 256, 256, 0, stream>>>(wo, wob, n3);
  }
  // 2) QKV GEMM: [4096,3072] = [4096,2048] x [3072,2048]^T
  gemm_bf16_wmma<<<dim3(QKV_N / BN, TOKENS / BM), 256, 0, stream>>>(
      hb, wqb, qkv32, TOKENS, QKV_N, HID);
  // 3) per-head LN + RoPE, scatter Q/K; V -> transposed bf16
  ln_rope_kernel<<<dim3(N_HEADS + N_KV, TOKENS), 64, 0, stream>>>(
      qkv32, pos, qlnw, klnw, qb, kb);
  {
    int nv = BATCH * N_KV * SEQ * HEAD_D;   // 2097152
    vcopy_kernel<<<(nv + 255) / 256, 256, 0, stream>>>(qkv32, vT);
  }
  // 4) flash attention
  attn_kernel<<<dim3(SEQ / 128, N_HEADS, BATCH), 256, 0, stream>>>(
      qb, kb, vT, ao);
  // 5) output projection: [4096,2048] = [4096,2048] x [2048,2048]^T
  gemm_bf16_wmma<<<dim3(HID / BN, TOKENS / BM), 256, 0, stream>>>(
      ao, wob, out, TOKENS, HID, HID);
}
